// MultiHeadAttentionQuantum_65481071398674
// MI455X (gfx1250) — compile-verified, matrix operand order checked by reading, not run on hardware
//
#include <hip/hip_runtime.h>

// MI455X / gfx1250 fused quantum-attention kernel.
// proj = closed-form quantum projection (prefix products of cosines),
// attention + output linear fused with v_wmma_f32_16x16x32_f16.

typedef __attribute__((ext_vector_type(16))) _Float16 v16h;
typedef __attribute__((ext_vector_type(8)))  _Float16 v8h;
typedef __attribute__((ext_vector_type(8)))  float    v8f;

#define B_   8
#define S_   1024
#define E_   64
#define H_   8
#define QT   64          // queries per workgroup
#define NTHR 128         // 4 waves (wave32)

// LDS layout (bytes)
#define OFF_K    0                         // [1024][64] f16 row-major
#define OFF_KT   (S_ * E_ * 2)             // [64][1024] f16 (transposed)
#define OFF_W    (OFF_KT + E_ * S_ * 2)    // [64][64] f16
#define OFF_ABUF (OFF_W + E_ * E_ * 2)     // 4 waves x [16][64] f16 staging
#define SMEM_BYTES (OFF_ABUF + 4 * 16 * E_ * 2)   // 278528 B < 320 KB

__global__ __launch_bounds__(NTHR)
void qattn_fused_kernel(const float* __restrict__ x,
                        const float* __restrict__ phi,
                        const float* __restrict__ W,
                        const float* __restrict__ bias,
                        float* __restrict__ out)
{
    extern __shared__ char smem[];
    _Float16* Ksh  = (_Float16*)(smem + OFF_K);
    _Float16* Kt   = (_Float16*)(smem + OFF_KT);
    _Float16* Wsh  = (_Float16*)(smem + OFF_W);
    _Float16* Abuf = (_Float16*)(smem + OFF_ABUF);

    const int tid   = threadIdx.x;        // 0..127
    const int wave  = tid >> 5;           // 0..3
    const int lane  = tid & 31;
    const int l15   = lane & 15;
    const int half  = lane >> 4;          // 0/1
    const int batch = blockIdx.y;
    const int qbase = blockIdx.x * QT;

    // ---------------- Phase 0: quantum projection into Ksh and Kt -------------
    // theta = x + phi per wire; <Z_0> = c1..c7 ; <Z_w> = c0..cw (w>=1)
    {
        const int h = tid & 7;            // head is fixed per thread
        float ph[8];
        #pragma unroll
        for (int w = 0; w < 8; ++w) ph[w] = phi[h * 8 + w];
        const float* xb = x + (size_t)batch * S_ * E_;
        for (int i = 0; i < 64; ++i) {            // 8192 (t,h) pairs / 128 thr
            const int t = (tid >> 3) + i * 16;
            const float* xr = xb + t * E_ + h * 8;
            __builtin_prefetch(xr + 16 * E_, 0, 0);   // global_prefetch_b8 (next tile)
            float c[8], f[8];
            #pragma unroll
            for (int w = 0; w < 8; ++w) c[w] = __cosf(xr[w] + ph[w]);
            f[0] = c[0];
            #pragma unroll
            for (int w = 1; w < 8; ++w) f[w] = f[w - 1] * c[w];
            float tail = c[1];
            #pragma unroll
            for (int w = 2; w < 8; ++w) tail *= c[w];
            v8h pv;
            pv[0] = (_Float16)tail;
            #pragma unroll
            for (int w = 1; w < 8; ++w) pv[w] = (_Float16)f[w];
            *(v8h*)(Ksh + t * E_ + h * 8) = pv;           // 16B ds_store_b128
            #pragma unroll
            for (int w = 0; w < 8; ++w) Kt[(h * 8 + w) * S_ + t] = pv[w];
        }
        for (int i = tid; i < E_ * E_; i += NTHR) Wsh[i] = (_Float16)W[i];
    }
    __syncthreads();

    // f16 A/B operand loader per CDNA5 16x32 layout:
    // lane(l): row = l&15 ; halfs[0..7]  = elems k = koff + (l>>4)*8 + 0..7
    //                       halfs[8..15] = elems k = koff + 16 + (l>>4)*8 + 0..7
    auto load_op = [&](const _Float16* base, int row, int stride, int koff) -> v16h {
        const _Float16* p = base + row * stride + koff + half * 8;
        v8h lo = *(const v8h*)(p);
        v8h hi = *(const v8h*)(p + 16);
        return __builtin_shufflevector(lo, hi, 0,1,2,3,4,5,6,7,8,9,10,11,12,13,14,15);
    };

    const float scale = 0.35355339059327373f;      // 1/sqrt(DK)
    const int qrow = qbase + wave * 16 + l15;
    v16h aq0 = load_op(Ksh, qrow, E_, 0);          // Q strip, k=0..31
    v16h aq1 = load_op(Ksh, qrow, E_, 32);         // Q strip, k=32..63

    // ---------------- Phase 1: row max of scaled scores ------------------------
    float rmax[8];
    #pragma unroll
    for (int j = 0; j < 8; ++j) rmax[j] = -3.0e38f;
    #pragma unroll 4
    for (int n = 0; n < 64; ++n) {                 // 64 key tiles of 16
        v16h b0 = load_op(Ksh, n * 16 + l15, E_, 0);
        v16h b1 = load_op(Ksh, n * 16 + l15, E_, 32);
        v8f c = {};
        c = __builtin_amdgcn_wmma_f32_16x16x32_f16(false, aq0, false, b0, (short)0, c, false, false);
        c = __builtin_amdgcn_wmma_f32_16x16x32_f16(false, aq1, false, b1, (short)0, c, false, false);
        #pragma unroll
        for (int j = 0; j < 8; ++j) rmax[j] = fmaxf(rmax[j], c[j]);
    }
    #pragma unroll
    for (int off = 1; off < 16; off <<= 1) {
        #pragma unroll
        for (int j = 0; j < 8; ++j)
            rmax[j] = fmaxf(rmax[j], __shfl_xor(rmax[j], off, 32));
    }
    #pragma unroll
    for (int j = 0; j < 8; ++j) rmax[j] *= scale;

    // ---------------- Phase 2: P = exp(s - max), ctx += P @ V ------------------
    _Float16* Pb = Abuf + wave * 16 * E_;          // per-wave staging [16][64]
    v8f acc[4] = {v8f{}, v8f{}, v8f{}, v8f{}};
    float rsum[8];
    #pragma unroll
    for (int j = 0; j < 8; ++j) rsum[j] = 0.0f;

    for (int kb2 = 0; kb2 < 16; ++kb2) {           // 64 keys per outer step
        #pragma unroll
        for (int u = 0; u < 2; ++u) {              // 2 independent 32-key chains
            const int kb = kb2 * 2 + u;            // (disjoint staging cols u*32)
            #pragma unroll
            for (int t2 = 0; t2 < 2; ++t2) {
                const int n = kb * 2 + t2;
                v16h b0 = load_op(Ksh, n * 16 + l15, E_, 0);
                v16h b1 = load_op(Ksh, n * 16 + l15, E_, 32);
                v8f c = {};
                c = __builtin_amdgcn_wmma_f32_16x16x32_f16(false, aq0, false, b0, (short)0, c, false, false);
                c = __builtin_amdgcn_wmma_f32_16x16x32_f16(false, aq1, false, b1, (short)0, c, false, false);
                #pragma unroll
                for (int j = 0; j < 8; ++j) {
                    float p = __expf(c[j] * scale - rmax[j]);   // p <= 1, f16-safe
                    rsum[j] += p;
                    // C-layout -> staging: row M=j+8*half, col u*32 + t2*16 + l15
                    Pb[(j + 8 * half) * E_ + u * 32 + t2 * 16 + l15] = (_Float16)p;
                }
            }
            // Same-wave DS ops are in-order: staging RAW is safe without barrier.
            v16h ap = load_op(Pb, l15, E_, u * 32); // 16x32 P tile as A operand
            #pragma unroll
            for (int e = 0; e < 4; ++e) {
                v16h bv = load_op(Kt, e * 16 + l15, S_, kb * 32);  // V^T rows
                acc[e] = __builtin_amdgcn_wmma_f32_16x16x32_f16(false, ap, false, bv, (short)0, acc[e], false, false);
            }
        }
    }

    // normalize by row sums
    #pragma unroll
    for (int off = 1; off < 16; off <<= 1) {
        #pragma unroll
        for (int j = 0; j < 8; ++j) rsum[j] += __shfl_xor(rsum[j], off, 32);
    }
    float rinv[8];
    #pragma unroll
    for (int j = 0; j < 8; ++j) rinv[j] = 1.0f / rsum[j];
    #pragma unroll
    for (int e = 0; e < 4; ++e) {
        #pragma unroll
        for (int j = 0; j < 8; ++j) acc[e][j] *= rinv[j];
    }

    // ---------------- Phase 3: out = ctx @ W^T + b -----------------------------
    #pragma unroll
    for (int e = 0; e < 4; ++e) {
        #pragma unroll
        for (int j = 0; j < 8; ++j)
            Pb[(j + 8 * half) * E_ + e * 16 + l15] = (_Float16)acc[e][j];
    }
    v16h a0 = load_op(Pb, l15, E_, 0);
    v16h a1 = load_op(Pb, l15, E_, 32);
    #pragma unroll
    for (int e = 0; e < 4; ++e) {
        v16h w0 = load_op(Wsh, e * 16 + l15, E_, 0);   // B col n = W row n
        v16h w1 = load_op(Wsh, e * 16 + l15, E_, 32);
        v8f o = {};
        o = __builtin_amdgcn_wmma_f32_16x16x32_f16(false, a0, false, w0, (short)0, o, false, false);
        o = __builtin_amdgcn_wmma_f32_16x16x32_f16(false, a1, false, w1, (short)0, o, false, false);
        const float be = bias[e * 16 + l15];
        float* orow = out + ((size_t)batch * S_ + qbase + wave * 16 + 8 * half) * E_ + e * 16 + l15;
        #pragma unroll
        for (int j = 0; j < 8; ++j) orow[j * E_] = o[j] + be;
    }
}

extern "C" void kernel_launch(void* const* d_in, const int* in_sizes, int n_in,
                              void* d_out, int out_size, void* d_ws, size_t ws_size,
                              hipStream_t stream) {
    const float* x    = (const float*)d_in[0];   // [8,1024,64]
    const float* phi  = (const float*)d_in[1];   // [8,8]
    const float* W    = (const float*)d_in[2];   // [64,64]
    const float* bias = (const float*)d_in[3];   // [64]
    float* out = (float*)d_out;                  // [8,1024,64]

    // 272KB dynamic LDS per workgroup (MI455X WGP has 320KB)
    (void)hipFuncSetAttribute((const void*)qattn_fused_kernel,
                              hipFuncAttributeMaxDynamicSharedMemorySize,
                              SMEM_BYTES);

    dim3 grid(S_ / QT, B_, 1);   // 16 query tiles x 8 batches = 128 WGs
    dim3 block(NTHR, 1, 1);      // 4 waves
    qattn_fused_kernel<<<grid, block, SMEM_BYTES, stream>>>(x, phi, W, bias, out);
}
